// Block_61701500174696
// MI455X (gfx1250) — matrix operand order repaired and checked by reading, not compiled, not run on hardware
//
#include <hip/hip_runtime.h>
#include <hip/hip_fp16.h>

#define DI __device__ __forceinline__

typedef __attribute__((ext_vector_type(16))) _Float16 v16h;
typedef __attribute__((ext_vector_type(8)))  float    v8f;

constexpr int kH = 64, kW = 64, kT = 32;
constexpr int kC = 96;
constexpr int kL = kH * kW * kT;       // 131072 tokens per batch image
constexpr int kM = 2 * kL;             // 262144 total tokens / window-tokens

struct F16F { union { v16h v; uint4 u[2]; }; };

DI v8f vzero() { v8f z = {0.f,0.f,0.f,0.f,0.f,0.f,0.f,0.f}; return z; }

DI v8f wmma16(v16h a, v16h b, v8f c) {
  // v_wmma_f32_16x16x32_f16: D = A(16x32 f16) * B(32x16 f16) + C(16x16 f32)
  return __builtin_amdgcn_wmma_f32_16x16x32_f16(false, a, false, b, (short)0, c, false, false);
}

// A fragment (16x32 f16), row-major source with stride `ld` halves.
// Contiguous k order used here: lanes 0-15 -> row=lane, k=0..15; lanes 16-31 -> row=lane-16, k=16..31
DI v16h frag_a(const _Float16* base, int ld) {
  int lane = (int)(threadIdx.x & 31);
  int r = lane & 15, hi = lane >> 4;
  const _Float16* p = base + r * ld + hi * 16;
  F16F f;
  f.u[0] = *(const uint4*)(p);
  f.u[1] = *(const uint4*)(p + 8);
  return f.v;
}

// B fragment (32x16 f16) from source stored [n][k] (row n, k contiguous, stride `ld` halves).
// Contiguous k order used here: lanes 0-15 -> n=lane, k=0..15; lanes 16-31 -> n=lane-16, k=16..31
DI v16h frag_b(const _Float16* base, int ld) {
  int lane = (int)(threadIdx.x & 31);
  int n = lane & 15, hi = lane >> 4;
  const _Float16* p = base + n * ld + hi * 16;
  F16F f;
  f.u[0] = *(const uint4*)(p);
  f.u[1] = *(const uint4*)(p + 8);
  return f.v;
}

DI float wred(float v) {
#pragma unroll
  for (int m = 16; m >= 1; m >>= 1) v += __shfl_xor(v, m, 32);
  return v;
}

// ---------------------------------------------------------------------------
// f32 -> f16 weight conversion
__global__ __launch_bounds__(256) void k_cvt(const float* __restrict__ in,
                                             _Float16* __restrict__ out, int n) {
  int i = blockIdx.x * 256 + threadIdx.x;
  if (i < n) out[i] = (_Float16)in[i];
}

// ---------------------------------------------------------------------------
// Fused: LayerNorm1 + shifted-window gather + QKV GEMM (WMMA) + head split.
// Grid: (kM/128, 3).  Block tile 128(tokens) x 96(cols of 288), K=96.
template <int SHIFT>
__global__ __launch_bounds__(256) void k_ln_qkv(
    const float* __restrict__ xsrc, const float* __restrict__ g1, const float* __restrict__ b1,
    const _Float16* __restrict__ w16 /*[288][96]*/, const float* __restrict__ bias /*[288]*/,
    _Float16* __restrict__ qbuf, _Float16* __restrict__ kbuf, _Float16* __restrict__ vbuf) {
  __shared__ __align__(16) _Float16 At[128 * 96];  // 24KB: LN'd tokens (f16)
  __shared__ __align__(16) _Float16 Bt[96 * 32];   // 6KB : weight k-chunk
  const int tid = threadIdx.x, lane = tid & 31, wv = tid >> 5;
  const int m0 = blockIdx.x * 128, n0 = blockIdx.y * 96;

  // --- LayerNorm + gather (roll by -SHIFT == read at +SHIFT) ---
  for (int t = 0; t < 16; ++t) {
    int tloc = wv * 16 + t;
    int token = m0 + tloc;
    int gwin = token >> 9, n = token & 511;
    int b = gwin >> 8, rem = gwin & 255;
    int hb = rem >> 5, wb = (rem >> 2) & 7, tb = rem & 3;
    int hh = (hb * 8 + (n >> 6) + SHIFT) & (kH - 1);
    int ww = (wb * 8 + ((n >> 3) & 7) + SHIFT) & (kW - 1);
    int tt = (tb * 8 + (n & 7) + SHIFT) & (kT - 1);
    const float* row = xsrc + ((size_t)b * kL + ((hh * kW + ww) * kT + tt)) * kC;
    float x0 = row[lane], x1 = row[lane + 32], x2 = row[lane + 64];
    float s = wred(x0 + x1 + x2);
    float sq = wred(x0 * x0 + x1 * x1 + x2 * x2);
    float mean = s * (1.0f / 96.0f);
    float inv = rsqrtf(sq * (1.0f / 96.0f) - mean * mean + 1e-5f);
    At[tloc * 96 + lane]      = (_Float16)((x0 - mean) * inv * g1[lane]      + b1[lane]);
    At[tloc * 96 + lane + 32] = (_Float16)((x1 - mean) * inv * g1[lane + 32] + b1[lane + 32]);
    At[tloc * 96 + lane + 64] = (_Float16)((x2 - mean) * inv * g1[lane + 64] + b1[lane + 64]);
  }
  __syncthreads();

  const int mw = wv >> 1, nw = wv & 1;  // wave tile: 32 x 48
  v8f acc[6];
#pragma unroll
  for (int i = 0; i < 6; ++i) acc[i] = vzero();

  for (int kc = 0; kc < 3; ++kc) {
#pragma unroll
    for (int it = 0; it < 2; ++it) {
      int idx = tid + it * 256;
      if (idx < 384) {
        int rn = idx >> 2, q = idx & 3;
        *(uint4*)&Bt[rn * 32 + q * 8] =
            *(const uint4*)&w16[(size_t)(n0 + rn) * 96 + kc * 32 + q * 8];
      }
    }
    __syncthreads();
    v16h af0 = frag_a(&At[(mw * 32 + 0) * 96 + kc * 32], 96);
    v16h af1 = frag_a(&At[(mw * 32 + 16) * 96 + kc * 32], 96);
    v16h bf0 = frag_b(&Bt[(nw * 48 + 0) * 32], 32);
    v16h bf1 = frag_b(&Bt[(nw * 48 + 16) * 32], 32);
    v16h bf2 = frag_b(&Bt[(nw * 48 + 32) * 32], 32);
    acc[0] = wmma16(af0, bf0, acc[0]); acc[1] = wmma16(af0, bf1, acc[1]);
    acc[2] = wmma16(af0, bf2, acc[2]); acc[3] = wmma16(af1, bf0, acc[3]);
    acc[4] = wmma16(af1, bf1, acc[4]); acc[5] = wmma16(af1, bf2, acc[5]);
    __syncthreads();
  }

  const int r15 = lane & 15, hi = lane >> 4;
#pragma unroll
  for (int tm = 0; tm < 2; ++tm) {
#pragma unroll
    for (int tn = 0; tn < 3; ++tn) {
      int ncol = n0 + nw * 48 + tn * 16 + r15;
      int which = ncol / 96;
      int c96 = ncol - which * 96;
      int head = c96 / 48, d = c96 - head * 48;
      float bv = bias[ncol];
      v8f a = acc[tm * 3 + tn];
#pragma unroll
      for (int rr = 0; rr < 8; ++rr) {
        int token = m0 + mw * 32 + tm * 16 + hi * 8 + rr;
        int gwin = token >> 9, n = token & 511;
        float val = a[rr] + bv;
        size_t widx = (size_t)gwin * 2 + head;
        if (which == 0)       qbuf[(widx * 512 + n) * 48 + d] = (_Float16)(val * 0.14433756729740643f);
        else if (which == 1)  kbuf[(widx * 512 + n) * 48 + d] = (_Float16)val;
        else                  vbuf[(widx * 48 + d) * 512 + n] = (_Float16)val;  // V stored d-major
      }
    }
  }
}

// ---------------------------------------------------------------------------
// Fused windowed attention: S=QK^T (+relpos bias, +shift mask), softmax, O=PV.
// Grid (32 row-blocks, 2 heads, 512 windows); 16 query rows per block.
template <int MASKED>
__global__ __launch_bounds__(256) void k_attn(
    const _Float16* __restrict__ qbuf, const _Float16* __restrict__ kbuf,
    const _Float16* __restrict__ vbuf, const float* __restrict__ rpb /*[3375][2]*/,
    _Float16* __restrict__ obuf /*[kM][96]*/) {
  __shared__ __align__(16) float    Sbuf[16 * 512];   // 32KB
  __shared__ __align__(16) _Float16 Pbuf[16 * 512];   // 16KB
  __shared__ float rpbl[3375];                        // 13.5KB (this head's column)
  __shared__ float rinv[16];
  __shared__ unsigned char rid[512];

  const int tid = threadIdx.x, lane = tid & 31, wv = tid >> 5;
  const int mb = blockIdx.x, head = blockIdx.y, gwin = blockIdx.z;
  const size_t widx = (size_t)gwin * 2 + head;
  const int r15 = lane & 15, hi = lane >> 4;

  for (int i = tid; i < 3375; i += 256) rpbl[i] = rpb[i * 2 + head];
  if (MASKED) {
    int rem = gwin & 255;
    int hb = rem >> 5, wb = (rem >> 2) & 7, tb = rem & 3;
    for (int n = tid; n < 512; n += 256) {
      int gh = hb * 8 + (n >> 6), gw = wb * 8 + ((n >> 3) & 7), gt = tb * 8 + (n & 7);
      int rh = gh < kH - 8 ? 0 : (gh < kH - 4 ? 1 : 2);
      int rw = gw < kW - 8 ? 0 : (gw < kW - 4 ? 1 : 2);
      int rt = gt < kT - 8 ? 0 : (gt < kT - 4 ? 1 : 2);
      rid[n] = (unsigned char)(rh * 9 + rw * 3 + rt);
    }
  }
  __syncthreads();

  // ---- S = Q K^T : hd=48 handled as k-chunks {0..31} and {32..47 | zero-pad} ----
  {
    const _Float16* qrow = qbuf + (widx * 512 + (size_t)mb * 16 + r15) * 48;
    F16F a0, a1;
    a0.u[0] = *(const uint4*)(qrow + hi * 16);
    a0.u[1] = *(const uint4*)(qrow + hi * 16 + 8);
    a1.u[0] = hi ? make_uint4(0, 0, 0, 0) : *(const uint4*)(qrow + 32);   // d 32..39 (hi=1: zero pad)
    a1.u[1] = hi ? make_uint4(0, 0, 0, 0) : *(const uint4*)(qrow + 40);   // d 40..47 (hi=1: zero pad)
#pragma unroll
    for (int jt = 0; jt < 4; ++jt) {
      int col0 = wv * 64 + jt * 16;
      const _Float16* krow = kbuf + (widx * 512 + (size_t)(col0 + r15)) * 48;
      F16F b0, b1;
      b0.u[0] = *(const uint4*)(krow + hi * 16);
      b0.u[1] = *(const uint4*)(krow + hi * 16 + 8);
      if (hi == 0) {  // k=32..47 real
        b1.u[0] = *(const uint4*)(krow + 32);
        b1.u[1] = *(const uint4*)(krow + 40);
      } else {        // k=48..63 zero pad
        b1.u[0] = make_uint4(0, 0, 0, 0);
        b1.u[1] = make_uint4(0, 0, 0, 0);
      }
      v8f acc = vzero();
      acc = wmma16(a0.v, b0.v, acc);
      acc = wmma16(a1.v, b1.v, acc);
#pragma unroll
      for (int rr = 0; rr < 8; ++rr) {
        int iloc = hi * 8 + rr;
        int i = mb * 16 + iloc;
        int j = col0 + r15;
        int dh = (i >> 6) - (j >> 6) + 7;
        int dw = ((i >> 3) & 7) - ((j >> 3) & 7) + 7;
        int dt = (i & 7) - (j & 7) + 7;
        float v = acc[rr] + rpbl[(dh * 15 + dw) * 15 + dt];
        if (MASKED) v += (rid[i] == rid[j]) ? 0.0f : -100.0f;
        Sbuf[iloc * 512 + j] = v;
      }
    }
  }
  __syncthreads();

  // ---- softmax over full rows (16 rows x 16 lanes x 32 cols) ----
  {
    int row = tid >> 4, sub = tid & 15;
    const float* srow = &Sbuf[row * 512 + sub * 32];
    _Float16* prow = &Pbuf[row * 512 + sub * 32];
    float mx = -1e30f;
    for (int c = 0; c < 32; ++c) mx = fmaxf(mx, srow[c]);
#pragma unroll
    for (int m = 1; m < 16; m <<= 1) mx = fmaxf(mx, __shfl_xor(mx, m, 32));
    float sum = 0.f;
    for (int c = 0; c < 32; ++c) {
      float e = __expf(srow[c] - mx);
      prow[c] = (_Float16)e;
      sum += e;
    }
#pragma unroll
    for (int m = 1; m < 16; m <<= 1) sum += __shfl_xor(sum, m, 32);
    if (sub == 0) rinv[row] = 1.0f / sum;
  }
  __syncthreads();

  // ---- O = P V : 16x48 output, waves 0..2 each own a 16-wide d-tile ----
  if (wv < 3) {
    int d0 = wv * 16;
    const _Float16* vrow = vbuf + (widx * 48 + (size_t)(d0 + r15)) * 512;
    v8f acc = vzero();
    for (int kc = 0; kc < 16; ++kc) {
      v16h a = frag_a(&Pbuf[kc * 32], 512);
      F16F bf;
      bf.u[0] = *(const uint4*)(vrow + kc * 32 + hi * 16);
      bf.u[1] = *(const uint4*)(vrow + kc * 32 + hi * 16 + 8);
      acc = wmma16(a, bf.v, acc);
    }
#pragma unroll
    for (int rr = 0; rr < 8; ++rr) {
      int row = hi * 8 + rr;
      size_t m = (size_t)gwin * 512 + mb * 16 + row;
      obuf[m * 96 + head * 48 + d0 + r15] = (_Float16)(acc[rr] * rinv[row]);
    }
  }
}

// ---------------------------------------------------------------------------
// Projection GEMM + window merge (unshift scatter) + residual.  Grid (kM/128).
template <int SHIFT>
__global__ __launch_bounds__(256) void k_proj(
    const _Float16* __restrict__ aobuf, const _Float16* __restrict__ w16 /*[96][96]*/,
    const float* __restrict__ bias, const float* __restrict__ shortcut,
    float* __restrict__ outx) {
  __shared__ __align__(16) _Float16 At[128 * 32];
  __shared__ __align__(16) _Float16 Bt[96 * 32];
  const int tid = threadIdx.x, lane = tid & 31, wv = tid >> 5;
  const int m0 = blockIdx.x * 128;
  const int mw = wv >> 1, nw = wv & 1;
  v8f acc[6];
#pragma unroll
  for (int i = 0; i < 6; ++i) acc[i] = vzero();

  for (int kc = 0; kc < 3; ++kc) {
#pragma unroll
    for (int it = 0; it < 2; ++it) {
      int idx = tid + it * 256;
      int rm = idx >> 2, q = idx & 3;
      *(uint4*)&At[rm * 32 + q * 8] =
          *(const uint4*)&aobuf[((size_t)(m0 + rm)) * 96 + kc * 32 + q * 8];
    }
#pragma unroll
    for (int it = 0; it < 2; ++it) {
      int idx = tid + it * 256;
      if (idx < 384) {
        int rn = idx >> 2, q = idx & 3;
        *(uint4*)&Bt[rn * 32 + q * 8] = *(const uint4*)&w16[(size_t)rn * 96 + kc * 32 + q * 8];
      }
    }
    __syncthreads();
    v16h af0 = frag_a(&At[(mw * 32 + 0) * 32], 32);
    v16h af1 = frag_a(&At[(mw * 32 + 16) * 32], 32);
    v16h bf0 = frag_b(&Bt[(nw * 48 + 0) * 32], 32);
    v16h bf1 = frag_b(&Bt[(nw * 48 + 16) * 32], 32);
    v16h bf2 = frag_b(&Bt[(nw * 48 + 32) * 32], 32);
    acc[0] = wmma16(af0, bf0, acc[0]); acc[1] = wmma16(af0, bf1, acc[1]);
    acc[2] = wmma16(af0, bf2, acc[2]); acc[3] = wmma16(af1, bf0, acc[3]);
    acc[4] = wmma16(af1, bf1, acc[4]); acc[5] = wmma16(af1, bf2, acc[5]);
    __syncthreads();
  }

  const int r15 = lane & 15, hi = lane >> 4;
#pragma unroll
  for (int tm = 0; tm < 2; ++tm) {
#pragma unroll
    for (int tn = 0; tn < 3; ++tn) {
      int ncol = nw * 48 + tn * 16 + r15;
      float bv = bias[ncol];
      v8f a = acc[tm * 3 + tn];
#pragma unroll
      for (int rr = 0; rr < 8; ++rr) {
        int token = m0 + mw * 32 + tm * 16 + hi * 8 + rr;
        int gwin = token >> 9, n = token & 511;
        int b = gwin >> 8, rem = gwin & 255;
        int hb = rem >> 5, wb = (rem >> 2) & 7, tb = rem & 3;
        int hh = (hb * 8 + (n >> 6) + SHIFT) & (kH - 1);
        int ww = (wb * 8 + ((n >> 3) & 7) + SHIFT) & (kW - 1);
        int tt = (tb * 8 + (n & 7) + SHIFT) & (kT - 1);
        size_t addr = ((size_t)b * kL + ((hh * kW + ww) * kT + tt)) * 96 + ncol;
        outx[addr] = shortcut[addr] + a[rr] + bv;
      }
    }
  }
}

// ---------------------------------------------------------------------------
// Fused MLP: LN2 -> f1 (WMMA) -> GELU (LDS) -> f2 (WMMA) -> residual, in place.
// Grid (kM/64); 64 tokens per block.
__global__ __launch_bounds__(256) void k_mlp(
    float* __restrict__ xbuf, const float* __restrict__ g2, const float* __restrict__ b2,
    const _Float16* __restrict__ w1 /*[384][96]*/, const float* __restrict__ bias1,
    const _Float16* __restrict__ w2 /*[96][384]*/, const float* __restrict__ bias2) {
  __shared__ __align__(16) _Float16 Ht[64 * 96];    // 12KB
  __shared__ __align__(16) _Float16 Gt[64 * 384];   // 48KB
  const int tid = threadIdx.x, lane = tid & 31, wv = tid >> 5;
  const size_t m0 = (size_t)blockIdx.x * 64;
  const int r15 = lane & 15, hi = lane >> 4;

  for (int t = 0; t < 8; ++t) {
    int tloc = wv * 8 + t;
    const float* row = xbuf + (m0 + tloc) * 96;
    float x0 = row[lane], x1 = row[lane + 32], x2 = row[lane + 64];
    float s = wred(x0 + x1 + x2);
    float sq = wred(x0 * x0 + x1 * x1 + x2 * x2);
    float mean = s * (1.0f / 96.0f);
    float inv = rsqrtf(sq * (1.0f / 96.0f) - mean * mean + 1e-5f);
    Ht[tloc * 96 + lane]      = (_Float16)((x0 - mean) * inv * g2[lane]      + b2[lane]);
    Ht[tloc * 96 + lane + 32] = (_Float16)((x1 - mean) * inv * g2[lane + 32] + b2[lane + 32]);
    Ht[tloc * 96 + lane + 64] = (_Float16)((x2 - mean) * inv * g2[lane + 64] + b2[lane + 64]);
  }
  __syncthreads();

  // stage 1: G(64x384) = gelu(H @ W1^T + b1); wave grid 2x4, wave tile 32x96
  {
    int mw = wv >> 2, nw = wv & 3;
    v8f acc[12];
#pragma unroll
    for (int i = 0; i < 12; ++i) acc[i] = vzero();
    for (int kc = 0; kc < 3; ++kc) {
      v16h af0 = frag_a(&Ht[(mw * 32 + 0) * 96 + kc * 32], 96);
      v16h af1 = frag_a(&Ht[(mw * 32 + 16) * 96 + kc * 32], 96);
#pragma unroll
      for (int tn = 0; tn < 6; ++tn) {
        v16h bf = frag_b(w1 + (size_t)(nw * 96 + tn * 16) * 96 + kc * 32, 96);
        acc[tn]     = wmma16(af0, bf, acc[tn]);
        acc[6 + tn] = wmma16(af1, bf, acc[6 + tn]);
      }
    }
#pragma unroll
    for (int tm = 0; tm < 2; ++tm) {
#pragma unroll
      for (int tn = 0; tn < 6; ++tn) {
        int ncol = nw * 96 + tn * 16 + r15;
        float bv = bias1[ncol];
        v8f a = acc[tm * 6 + tn];
#pragma unroll
        for (int rr = 0; rr < 8; ++rr) {
          float v = a[rr] + bv;
          v = 0.5f * v * (1.0f + erff(v * 0.7071067811865476f));
          Gt[(mw * 32 + tm * 16 + hi * 8 + rr) * 384 + ncol] = (_Float16)v;
        }
      }
    }
  }
  __syncthreads();

  // stage 2: Y(64x96) = G @ W2^T + b2 + x; wave grid 4x2, wave tile 16x48
  {
    int mw = wv >> 1, nw = wv & 1;
    v8f acc[3];
#pragma unroll
    for (int i = 0; i < 3; ++i) acc[i] = vzero();
    for (int kc = 0; kc < 12; ++kc) {
      v16h a = frag_a(&Gt[(mw * 16) * 384 + kc * 32], 384);
#pragma unroll
      for (int tn = 0; tn < 3; ++tn) {
        v16h bf = frag_b(w2 + (size_t)(nw * 48 + tn * 16) * 384 + kc * 32, 384);
        acc[tn] = wmma16(a, bf, acc[tn]);
      }
    }
#pragma unroll
    for (int tn = 0; tn < 3; ++tn) {
      int ncol = nw * 48 + tn * 16 + r15;
      float bv = bias2[ncol];
#pragma unroll
      for (int rr = 0; rr < 8; ++rr) {
        size_t token = m0 + mw * 16 + hi * 8 + rr;
        size_t addr = token * 96 + ncol;
        xbuf[addr] += acc[tn][rr] + bv;
      }
    }
  }
}

// ---------------------------------------------------------------------------
// Final LN + transpose to (B, C, H, W, T).  Grid (kM/8), one token per wave.
__global__ __launch_bounds__(256) void k_final(const float* __restrict__ xbuf,
                                               const float* __restrict__ g,
                                               const float* __restrict__ bb,
                                               float* __restrict__ out) {
  const int tid = threadIdx.x, lane = tid & 31, wv = tid >> 5;
  size_t token = (size_t)blockIdx.x * 8 + wv;
  const float* row = xbuf + token * 96;
  float x0 = row[lane], x1 = row[lane + 32], x2 = row[lane + 64];
  float s = wred(x0 + x1 + x2);
  float sq = wred(x0 * x0 + x1 * x1 + x2 * x2);
  float mean = s * (1.0f / 96.0f);
  float inv = rsqrtf(sq * (1.0f / 96.0f) - mean * mean + 1e-5f);
  int b = (int)(token >> 17);
  size_t l = token & (size_t)(kL - 1);
  out[((size_t)b * 96 + lane) * kL + l]        = (x0 - mean) * inv * g[lane]      + bb[lane];
  out[((size_t)b * 96 + lane + 32) * kL + l]   = (x1 - mean) * inv * g[lane + 32] + bb[lane + 32];
  out[((size_t)b * 96 + lane + 64) * kL + l]   = (x2 - mean) * inv * g[lane + 64] + bb[lane + 64];
}

// ---------------------------------------------------------------------------
extern "C" void kernel_launch(void* const* d_in, const int* in_sizes, int n_in,
                              void* d_out, int out_size, void* d_ws, size_t ws_size,
                              hipStream_t stream) {
  (void)in_sizes; (void)n_in; (void)out_size; (void)ws_size;
  const float* x = (const float*)d_in[0];
  const float* wP[13]; const float* sP[13];
  for (int i = 0; i < 13; ++i) { wP[i] = (const float*)d_in[4 + i]; sP[i] = (const float*)d_in[17 + i]; }
  // 0 n1g,1 n1b,2 qkvw,3 qkvb,4 projw,5 projb,6 rpb,7 n2g,8 n2b,9 f1w,10 f1b,11 f2w,12 f2b
  const float* ng = (const float*)d_in[30];
  const float* nb = (const float*)d_in[31];

  char* ws = (char*)d_ws;
  _Float16* w16 = (_Float16*)ws;           // f16 weight region (half offsets below)
  constexpr size_t oQKVW = 0, oPROJW = 27648, oF1W = 36864, oF2W = 73728, wStride = 110592;
  constexpr size_t SZ_QKVBUF = (size_t)512 * 2 * 512 * 48 * sizeof(_Float16);  // 48MB
  constexpr size_t OFF_Q  = 458752;
  constexpr size_t OFF_K  = OFF_Q + SZ_QKVBUF;
  constexpr size_t OFF_V  = OFF_K + SZ_QKVBUF;
  constexpr size_t OFF_AO = OFF_V + SZ_QKVBUF;
  constexpr size_t OFF_X  = OFF_AO + (size_t)kM * 96 * sizeof(_Float16);
  _Float16* qbuf  = (_Float16*)(ws + OFF_Q);
  _Float16* kbuf  = (_Float16*)(ws + OFF_K);
  _Float16* vbuf  = (_Float16*)(ws + OFF_V);
  _Float16* aobuf = (_Float16*)(ws + OFF_AO);
  float*    xbuf  = (float*)(ws + OFF_X);

  auto cvt = [&](const float* src, size_t off_h, int n) {
    k_cvt<<<(n + 255) / 256, 256, 0, stream>>>(src, w16 + off_h, n);
  };
  cvt(wP[2], oQKVW, 288 * 96);            cvt(wP[4], oPROJW, 96 * 96);
  cvt(wP[9], oF1W, 384 * 96);             cvt(wP[11], oF2W, 384 * 96);
  cvt(sP[2], wStride + oQKVW, 288 * 96);  cvt(sP[4], wStride + oPROJW, 96 * 96);
  cvt(sP[9], wStride + oF1W, 384 * 96);   cvt(sP[11], wStride + oF2W, 384 * 96);

  dim3 gq(kM / 128, 3), gat(32, 2, 512);
  // ---- block 1: W-MSA (shift 0, no mask) ----
  k_ln_qkv<0><<<gq, 256, 0, stream>>>(x, wP[0], wP[1], w16 + oQKVW, wP[3], qbuf, kbuf, vbuf);
  k_attn<0><<<gat, 256, 0, stream>>>(qbuf, kbuf, vbuf, wP[6], aobuf);
  k_proj<0><<<kM / 128, 256, 0, stream>>>(aobuf, w16 + oPROJW, wP[5], x, xbuf);
  k_mlp<<<kM / 64, 256, 0, stream>>>(xbuf, wP[7], wP[8], w16 + oF1W, wP[10], w16 + oF2W, wP[12]);
  // ---- block 2: SW-MSA (shift 4, masked) ----
  k_ln_qkv<4><<<gq, 256, 0, stream>>>(xbuf, sP[0], sP[1], w16 + wStride + oQKVW, sP[3], qbuf, kbuf, vbuf);
  k_attn<1><<<gat, 256, 0, stream>>>(qbuf, kbuf, vbuf, sP[6], aobuf);
  k_proj<4><<<kM / 128, 256, 0, stream>>>(aobuf, w16 + wStride + oPROJW, sP[5], xbuf, xbuf);
  k_mlp<<<kM / 64, 256, 0, stream>>>(xbuf, sP[7], sP[8], w16 + wStride + oF1W, sP[10],
                                     w16 + wStride + oF2W, sP[12]);
  // ---- final norm + transpose ----
  k_final<<<kM / 8, 256, 0, stream>>>(xbuf, ng, nb, (float*)d_out);
}
